// Seq2Seq_58402965291111
// MI455X (gfx1250) — matrix operand order repaired and checked by reading, not compiled, not run on hardware
//
#include <hip/hip_runtime.h>

typedef __attribute__((ext_vector_type(16))) _Float16 v16h;
typedef __attribute__((ext_vector_type(8)))  float    v8f;

#define H_     256
#define NJ_    22
#define EMB_   300
#define L_IN_  1024
#define L_OUT_ 2048
#define VPAD_  352   // f16 activation buffers: up to K=320 valid + 32-half zero pad

// ---------------------------------------------------------------------------
// Pack a GRU cell's [Wi ; Wh] (each 768 x K) into ONE 96-tile WMMA A-blob,
// zero-padded to KB*32 columns. Tiles 0..47 = Wi, 48..95 = Wh. Fragment for
// (tile T, kblock kb, lane) is 16 contiguous halves at ((T*KB+kb)*32+lane)*16
// -> the per-lane A load is one contiguous 32-byte global load.
// A layout (16x32 f16): lanes 0-15 -> M=lane, element j holds K=j;
// lanes 16-31: same M, K=16+j (contiguous, matches B fill).
// ---------------------------------------------------------------------------
__global__ void pack_pair_kernel(const float* __restrict__ Wi, const float* __restrict__ Wh,
                                 _Float16* __restrict__ dst, int K_wi, int K_wh, int KB) {
  int idx = blockIdx.x * blockDim.x + threadIdx.x;
  int total = 96 * KB * 512;
  if (idx >= total) return;
  int j    = idx & 15;
  int lane = (idx >> 4) & 31;
  int tk   = idx >> 9;            // T*KB + kb
  int kb   = tk % KB;
  int T    = tk / KB;
  int koff = ((lane >= 16) ? 16 : 0) + j;
  int k    = kb * 32 + koff;
  const float* W = (T < 48) ? Wi : Wh;
  int K_in = (T < 48) ? K_wi : K_wh;
  int t48  = (T < 48) ? T : T - 48;
  int row  = t48 * 16 + (lane & 15);
  float v = (k < K_in) ? W[row * K_in + k] : 0.0f;
  dst[idx] = (_Float16)v;
}

// ---------------------------------------------------------------------------
// One fused GRU-gate pass: gi = Wi@x, gh = Wh@h in a single 96-tile WMMA
// sweep. B operand carries x in column 0 (lanes 0/16) and h in column 1
// (lanes 1/17); lanes with col>=2 load from the zero pad of vx (no selects).
// b[KB] stays register-resident for the whole pass; A fragments stream from
// global through a depth-4 prefetch ring (~120 VGPRs total -> no spills).
// ---------------------------------------------------------------------------
template<int KB>
__device__ __forceinline__ void gates_wmma(const _Float16* __restrict__ Wp,
                                           const _Float16* __restrict__ vx,
                                           const _Float16* __restrict__ vh,
                                           float* __restrict__ gi, float* __restrict__ gh,
                                           int tid) {
  const int wave  = tid >> 5;
  const int lane  = tid & 31;
  const int col   = lane & 15;
  const int khalf = (lane >> 4) << 4;           // 0 (K 0-15) or 16 (K 16-31)
  const int zoff  = KB * 32;                    // zero-pad region
  const _Float16* src = (col == 1) ? vh : vx;

  v16h b[KB];
#pragma unroll
  for (int kb = 0; kb < KB; ++kb) {
    int off = (col <= 1) ? (kb * 32 + khalf) : zoff;
    b[kb] = *(const v16h*)(src + off);
  }

  const _Float16* base = Wp + (wave * KB) * 512 + lane * 16;
  const int NF = 12 * KB;                       // flattened (tile, kblock) stream
  const v8f vzero = {};

  v16h a[4];
#pragma unroll
  for (int p = 0; p < 4; ++p)                   // p < 4 <= KB -> tile 0, kb=p
    a[p] = *(const v16h*)(base + p * 512);

  v8f acc = vzero;
#pragma unroll
  for (int f = 0; f < NF; ++f) {
    int kb = f % KB;
    acc = __builtin_amdgcn_wmma_f32_16x16x32_f16(false, a[f & 3], false, b[kb],
                                                 (short)0, acc, false, false);
    int fn = f + 4;
    if (fn < NF) {                              // refill the slot just consumed
      int itn = fn / KB, kbn = fn % KB;
      a[f & 3] = *(const v16h*)(base + (itn * 8 * KB + kbn) * 512);
    }
    if (kb == KB - 1) {                         // tile boundary: write column out
      int it = f / KB;
      const bool isWi = (it < 6);               // tiles 0..47 = Wi, 48..95 = Wh
      float* y = isWi ? gi : gh;
      int T = wave + 8 * it;
      int t = isWi ? T : T - 48;
      if (col == (isWi ? 0 : 1)) {
        int basey = t * 16 + ((lane >= 16) ? 8 : 0);
#pragma unroll
        for (int r2 = 0; r2 < 8; ++r2) y[basey + r2] = acc[r2];
      }
      acc = vzero;
    }
  }
}

__device__ __forceinline__ float sigmoidf_(float x) { return 1.0f / (1.0f + __expf(-x)); }

// GRU epilogue; thread i updates h[i] (f32 carry) and hh[i] (f16 WMMA feed).
__device__ __forceinline__ void gru_combine(const float* gi, const float* gh,
                                            const float* __restrict__ bi,
                                            const float* __restrict__ bh,
                                            float* h, _Float16* hh, int i) {
  float rr  = sigmoidf_((gi[i]      + bi[i])      + (gh[i]      + bh[i]));
  float zz  = sigmoidf_((gi[H_+i]   + bi[H_+i])   + (gh[H_+i]   + bh[H_+i]));
  float inn =  gi[2*H_+i] + bi[2*H_+i];
  float hn  =  gh[2*H_+i] + bh[2*H_+i];
  float nn  = tanhf(inn + rr * hn);
  float hv  = (1.0f - zz) * nn + zz * h[i];
  h[i]  = hv;
  hh[i] = (_Float16)hv;
}

__device__ __forceinline__ float wave_red_add(float v) {
#pragma unroll
  for (int m = 16; m >= 1; m >>= 1) v += __shfl_xor(v, m, 32);
  return v;
}
__device__ __forceinline__ float block_red_add(float v, float* red, int tid) {
  v = wave_red_add(v);
  if ((tid & 31) == 0) red[tid >> 5] = v;
  __syncthreads();
  float r = 0.0f;
#pragma unroll
  for (int w = 0; w < 8; ++w) r += red[w];
  __syncthreads();
  return r;
}
__device__ __forceinline__ void block_red_add2(float a, float b, float* red, int tid,
                                               float& ra, float& rb) {
  a = wave_red_add(a);
  b = wave_red_add(b);
  if ((tid & 31) == 0) { red[tid >> 5] = a; red[8 + (tid >> 5)] = b; }
  __syncthreads();
  ra = 0.0f; rb = 0.0f;
#pragma unroll
  for (int w = 0; w < 8; ++w) { ra += red[w]; rb += red[8 + w]; }
  __syncthreads();
}

// ---------------------------------------------------------------------------
// Encoder: 1024 sequential steps, single workgroup (8 waves).
// ---------------------------------------------------------------------------
__global__ void __launch_bounds__(256)
encoder_kernel(const float* __restrict__ word_embs,
               const _Float16* __restrict__ pL0, const _Float16* __restrict__ pL1,
               const float* __restrict__ bi0, const float* __restrict__ bh0,
               const float* __restrict__ bi1, const float* __restrict__ bh1,
               float* __restrict__ out0, float* __restrict__ hfinal) {
  __shared__ _Float16 xh[VPAD_], h0h[VPAD_], h1h[VPAD_];
  __shared__ float h0[H_], h1[H_], gi[3*H_], gh[3*H_];
  int tid = threadIdx.x;
  for (int j = tid; j < VPAD_; j += 256) { xh[j] = (_Float16)0.0f; h0h[j] = (_Float16)0.0f; h1h[j] = (_Float16)0.0f; }
  h0[tid] = 0.0f; h1[tid] = 0.0f;
  __syncthreads();

  for (int step = 0; step < L_IN_; ++step) {
    for (int j = tid; j < EMB_; j += 256) xh[j] = (_Float16)word_embs[step * EMB_ + j];
    __syncthreads();
    gates_wmma<10>(pL0, xh, h0h, gi, gh, tid);   // Wi0@x (K=300 pad 320), Wh0@h0 (pad 320)
    __syncthreads();
    gru_combine(gi, gh, bi0, bh0, h0, h0h, tid);
    __syncthreads();
    gates_wmma<8>(pL1, h0h, h1h, gi, gh, tid);   // Wi1@h0, Wh1@h1
    __syncthreads();
    gru_combine(gi, gh, bi1, bh1, h1, h1h, tid);
    __syncthreads();
    if (tid == 0) out0[step] = h1[0];            // enc_os row value (bug-faithful)
  }
  __syncthreads();
  hfinal[tid] = h0[tid];
  hfinal[H_ + tid] = h1[tid];
}

// ---------------------------------------------------------------------------
// Decoder: 2048 sequential steps. Attention collapses to a scalar s (enc_os
// rows are constant); tanh scores are in (-1,1) so softmax needs no max-shift.
// ---------------------------------------------------------------------------
__global__ void __launch_bounds__(256)
decoder_kernel(const float* __restrict__ expected_frames, const float* __restrict__ out0,
               const _Float16* __restrict__ pL0, const _Float16* __restrict__ pL1,
               const float* __restrict__ bi0, const float* __restrict__ bh0,
               const float* __restrict__ bi1, const float* __restrict__ bh1,
               const float* __restrict__ attn_W, const float* __restrict__ attn_b,
               const float* __restrict__ comb_W, const float* __restrict__ comb_b,
               const float* __restrict__ out_W,  const float* __restrict__ out_b,
               const float* __restrict__ hinit,  float* __restrict__ preds) {
  __shared__ _Float16 xh[VPAD_], h0h[VPAD_], h1h[VPAD_];
  __shared__ float h0[H_], h1[H_], gi[3*H_], gh[3*H_];
  __shared__ float din[NJ_], red[16], out0s[L_IN_], combRS[H_], attR[H_];
  int tid = threadIdx.x;

  for (int j = tid; j < VPAD_; j += 256) { xh[j] = (_Float16)0.0f; h0h[j] = (_Float16)0.0f; h1h[j] = (_Float16)0.0f; }
  float h0v = hinit[tid], h1v = hinit[H_ + tid];
  h0[tid] = h0v; h1[tid] = h1v;
  h0h[tid] = (_Float16)h0v; h1h[tid] = (_Float16)h1v;
  if (tid < NJ_) din[tid] = expected_frames[tid];
  for (int i = tid; i < L_IN_; i += 256) out0s[i] = out0[i];
  attR[tid] = attn_W[tid];                       // attn_W[0, :256]
  {
    float cs = 0.0f;
    const float* cw = comb_W + tid * (H_ + NJ_);
    for (int j = NJ_; j < H_ + NJ_; ++j) cs += cw[j];
    combRS[tid] = cs;                            // rowsum(comb_W[:, 22:278])
  }
  __syncthreads();
  float c0 = block_red_add(attn_W[H_ + tid], red, tid);   // sum attn_W[0, 256:512]

  for (int step = 0; step < L_OUT_; ++step) {
    // a0 = attn_W[0,:256] . h1 + attn_b[0]
    float a0 = block_red_add(attR[tid] * h1[tid], red, tid) + attn_b[0];

    // softmax over encoder positions, channel 0; ctx scalar s
    float pe = 0.0f, pn = 0.0f;
    for (int i = tid; i < L_IN_; i += 256) {
      float o = out0s[i];
      float e = __expf(tanhf(a0 + o * c0));      // tanh in (-1,1): exp safe
      pe += e; pn += e * o;
    }
    float den, num;
    block_red_add2(pe, pn, red, tid, den, num);
    float s = num / den;

    // x = relu(comb_W[:,:22] @ din + s*rowsum + comb_b)  (f16 for WMMA feed)
    {
      float acc = comb_b[tid] + s * combRS[tid];
      const float* cw = comb_W + tid * (H_ + NJ_);
#pragma unroll
      for (int j = 0; j < NJ_; ++j) acc += cw[j] * din[j];
      xh[tid] = (_Float16)fmaxf(acc, 0.0f);
    }
    __syncthreads();

    gates_wmma<8>(pL0, xh, h0h, gi, gh, tid);
    __syncthreads();
    gru_combine(gi, gh, bi0, bh0, h0, h0h, tid);
    __syncthreads();
    gates_wmma<8>(pL1, h0h, h1h, gi, gh, tid);
    __syncthreads();
    gru_combine(gi, gh, bi1, bh1, h1, h1h, tid);
    __syncthreads();

    // pred = out_W @ h1 + out_b : 8 lanes per row + shfl reduce
    {
      int row = tid >> 3, sub = tid & 7;
      float pa = 0.0f;
      if (row < NJ_) {
        const float* ow = out_W + row * H_;
        for (int jj = sub; jj < H_; jj += 8) pa += ow[jj] * h1[jj];
      }
      pa += __shfl_xor(pa, 1, 32);
      pa += __shfl_xor(pa, 2, 32);
      pa += __shfl_xor(pa, 4, 32);
      if (row < NJ_ && sub == 0) {
        float p = pa + out_b[row];
        preds[step * NJ_ + row] = p;
        din[row] = p;                            // next decoder input
      }
    }
    __syncthreads();
  }
}

extern "C" void kernel_launch(void* const* d_in, const int* in_sizes, int n_in,
                              void* d_out, int out_size, void* d_ws, size_t ws_size,
                              hipStream_t stream) {
  (void)in_sizes; (void)n_in; (void)out_size; (void)ws_size;
  const float* word_embs       = (const float*)d_in[0];
  const float* expected_frames = (const float*)d_in[1];
  const float* enc_Wi0 = (const float*)d_in[2];
  const float* enc_Wh0 = (const float*)d_in[3];
  const float* enc_bi0 = (const float*)d_in[4];
  const float* enc_bh0 = (const float*)d_in[5];
  const float* enc_Wi1 = (const float*)d_in[6];
  const float* enc_Wh1 = (const float*)d_in[7];
  const float* enc_bi1 = (const float*)d_in[8];
  const float* enc_bh1 = (const float*)d_in[9];
  const float* dec_Wi0 = (const float*)d_in[10];
  const float* dec_Wh0 = (const float*)d_in[11];
  const float* dec_bi0 = (const float*)d_in[12];
  const float* dec_bh0 = (const float*)d_in[13];
  const float* dec_Wi1 = (const float*)d_in[14];
  const float* dec_Wh1 = (const float*)d_in[15];
  const float* dec_bi1 = (const float*)d_in[16];
  const float* dec_bh1 = (const float*)d_in[17];
  const float* attn_W  = (const float*)d_in[18];
  const float* attn_b  = (const float*)d_in[19];
  const float* comb_W  = (const float*)d_in[20];
  const float* comb_b  = (const float*)d_in[21];
  const float* out_W   = (const float*)d_in[22];
  const float* out_b   = (const float*)d_in[23];
  float* preds = (float*)d_out;

  // ws: 4 fused A-blobs (f16), then out0[1024] + hfinal[512] (f32)
  const size_t PENC0 = (size_t)96 * 10 * 512;   // 491,520 halves (KB=10)
  const size_t P256  = (size_t)96 * 8  * 512;   // 393,216 halves (KB=8)
  _Float16* pEncL0 = (_Float16*)d_ws;
  _Float16* pEncL1 = pEncL0 + PENC0;
  _Float16* pDecL0 = pEncL1 + P256;
  _Float16* pDecL1 = pDecL0 + P256;
  float* out0   = (float*)(pDecL1 + P256);
  float* hfinal = out0 + L_IN_;

  dim3 blk(256);
  pack_pair_kernel<<<(unsigned)((PENC0 + 255) / 256), blk, 0, stream>>>(enc_Wi0, enc_Wh0, pEncL0, EMB_, H_, 10);
  pack_pair_kernel<<<(unsigned)((P256  + 255) / 256), blk, 0, stream>>>(enc_Wi1, enc_Wh1, pEncL1, H_,   H_, 8);
  pack_pair_kernel<<<(unsigned)((P256  + 255) / 256), blk, 0, stream>>>(dec_Wi0, dec_Wh0, pDecL0, H_,   H_, 8);
  pack_pair_kernel<<<(unsigned)((P256  + 255) / 256), blk, 0, stream>>>(dec_Wi1, dec_Wh1, pDecL1, H_,   H_, 8);

  encoder_kernel<<<1, blk, 0, stream>>>(word_embs, pEncL0, pEncL1,
      enc_bi0, enc_bh0, enc_bi1, enc_bh1, out0, hfinal);

  decoder_kernel<<<1, blk, 0, stream>>>(expected_frames, out0, pDecL0, pDecL1,
      dec_bi0, dec_bh0, dec_bi1, dec_bh1,
      attn_W, attn_b, comb_W, comb_b, out_W, out_b,
      hfinal, preds);
}
